// MoRALayer_64080912056585
// MI455X (gfx1250) — matrix operand order repaired and checked by reading, not compiled, not run on hardware
//
#include <hip/hip_runtime.h>
#include <hip/hip_bf16.h>

// ---------------------------------------------------------------------------
// MoRA layer for MI455X (gfx1250): ACT-LSTM + 2-level top-2 MoE + LayerNorm.
// GEMMs: v_wmma_f32_16x16x32_bf16, double-buffered LDS staging fed by
// global_load_async_to_lds_b128 (ASYNCcnt), one barrier per K-step.
// Workspace requirement: ~1 GB (see carve list in kernel_launch).
// ---------------------------------------------------------------------------

#define BB     4096          // batch / tokens
#define DD     2048          // feature dim
#define FD     8192          // 4*D (LSTM gates)
#define NE     8             // experts per level
#define NL     2             // levels
#define NSTEP  8             // ACT max steps
#define KTOP   2             // top-k
#define CAP    (BB*KTOP + NE*128)   // 9216: padded grouped-GEMM row capacity

#define BM 128               // block tile M
#define BN 128               // block tile N
#define BK 32                // K per stage (one WMMA K-step)
#define KPAD 56              // LDS row stride in halves: 112B (16B-aligned)

#define USE_ASYNC_LDS 1      // global_load_async_to_lds_b128 staging path

#define GF_BIAS   1
#define GF_BIAS2  2
#define GF_ADDMAT 4
#define GF_RELU   8

typedef __attribute__((ext_vector_type(16))) __bf16 v16bf;
typedef __attribute__((ext_vector_type(8)))  float  v8f;

union FragU { v16bf v; uint4 q[2]; };

__device__ __forceinline__ unsigned short f32_to_bf16(float f) {
  unsigned int u = __float_as_uint(f);
  u += 0x7FFFu + ((u >> 16) & 1u);      // round-to-nearest-even
  return (unsigned short)(u >> 16);
}

__device__ __forceinline__ float sigmf(float x) { return 1.0f / (1.0f + __expf(-x)); }

#if USE_ASYNC_LDS
// vdst = 32-bit LDS byte address, vaddr = 64-bit global address (GV mode)
#define ASYNC_LOAD_B128(ldsaddr, gaddr)                                        \
  asm volatile("global_load_async_to_lds_b128 %0, %1, off"                     \
               :: "v"(ldsaddr), "v"(gaddr) : "memory")
#define WAIT_ASYNC() asm volatile("s_wait_asynccnt 0" ::: "memory")
#endif

// ---------------------------------------------------------------------------
// fp32 -> bf16 bulk conversion
// ---------------------------------------------------------------------------
__global__ void cvt_bf16_kernel(const float* __restrict__ src,
                                unsigned short* __restrict__ dst, size_t n) {
  size_t i = (size_t)blockIdx.x * blockDim.x + threadIdx.x;
  size_t stride = (size_t)gridDim.x * blockDim.x;
  for (; i < n; i += stride) dst[i] = f32_to_bf16(src[i]);
}

// ---------------------------------------------------------------------------
// WMMA GEMM: out[M,N](f32 and/or bf16) = epi( A[M,K](bf16) @ W[N,K](bf16)^T )
// Block tile 128x128, 8 waves (4 along M x 2 along N), wave tile 32x64.
// Double-buffered LDS; async-to-LDS staging for the next K-slab overlaps the
// WMMA work on the current one.  Optionally grouped over experts via
// 128-aligned row offsets (offs[NE+1]).
// ---------------------------------------------------------------------------
__global__ __launch_bounds__(256)
void gemm_bf16_kernel(const unsigned short* __restrict__ A, int lda,
                      const unsigned short* __restrict__ W,
                      float* __restrict__ outf,
                      unsigned short* __restrict__ outbf, int ldc,
                      const float* __restrict__ bias,
                      const float* __restrict__ bias2,
                      const float* __restrict__ addmat,
                      int Kdim, int flags,
                      const int* __restrict__ activeFlag,
                      const int* __restrict__ offs,
                      int wStride, int biasStride) {
  if (activeFlag && *activeFlag == 0) return;

  const int m0 = blockIdx.x * BM;
  const int n0 = blockIdx.y * BN;

  const unsigned short* Wp = W;
  const float* biasp = bias;
  if (offs) {                                   // grouped (MoE) mode
    if (m0 >= offs[NE]) return;                 // beyond last expert segment
    int e = 0;
    while (e + 1 < NE && m0 >= offs[e + 1]) ++e;
    Wp = W + (size_t)e * wStride;
    if (bias) biasp = bias + (size_t)e * biasStride;
  }

  __shared__ unsigned short As[2][BM][KPAD];
  __shared__ unsigned short Ws[2][BN][KPAD];

  const int tid  = threadIdx.x;
  const int lane = tid & 31;
  const int wid  = tid >> 5;     // 0..7
  const int wm   = wid & 3;      // 4 waves along M (32 rows each)
  const int wn   = wid >> 2;     // 2 waves along N (64 cols each)
  const int half = lane >> 4;    // 0/1
  const int l16  = lane & 15;

  // ---- per-thread staging map: 2 A-chunks + 2 W-chunks of 16B per K-slab ----
  // A tile: 128 rows x 32 halves = 512 chunks; same for W tile.
  const unsigned short* aSrc[2];
  const unsigned short* wSrc[2];
  unsigned aDst[2], wDst[2];
#pragma unroll
  for (int i = 0; i < 2; ++i) {
    int ch = tid + i * 256;
    int r = ch >> 2, c = ch & 3;
    aSrc[i] = A  + (size_t)(m0 + r) * lda  + c * 8;
    wSrc[i] = Wp + (size_t)(n0 + r) * Kdim + c * 8;
    aDst[i] = (unsigned)(uintptr_t)(void*)&As[0][r][c * 8];
    wDst[i] = (unsigned)(uintptr_t)(void*)&Ws[0][r][c * 8];
  }
  const unsigned aBufBytes = (unsigned)sizeof(As[0]);
  const unsigned wBufBytes = (unsigned)sizeof(Ws[0]);

  auto stage = [&](int buf, int k0) {
#if USE_ASYNC_LDS
#pragma unroll
    for (int i = 0; i < 2; ++i) {
      ASYNC_LOAD_B128(aDst[i] + (unsigned)buf * aBufBytes, (const void*)(aSrc[i] + k0));
      ASYNC_LOAD_B128(wDst[i] + (unsigned)buf * wBufBytes, (const void*)(wSrc[i] + k0));
    }
#else
#pragma unroll
    for (int i = 0; i < 2; ++i) {
      int ch = tid + i * 256;
      int r = ch >> 2, c = ch & 3;
      *(uint4*)&As[buf][r][c * 8] = *(const uint4*)(aSrc[i] + k0);
      *(uint4*)&Ws[buf][r][c * 8] = *(const uint4*)(wSrc[i] + k0);
    }
#endif
  };

  v8f acc[2][4];
#pragma unroll
  for (int i = 0; i < 2; ++i)
#pragma unroll
    for (int j = 0; j < 4; ++j)
#pragma unroll
      for (int r = 0; r < 8; ++r) acc[i][j][r] = 0.0f;

  // ---- prologue: stage first slab ----
  stage(0, 0);
#if USE_ASYNC_LDS
  WAIT_ASYNC();
#endif
  __syncthreads();

  for (int k0 = 0; k0 < Kdim; k0 += BK) {
    const int cur = (k0 / BK) & 1;
    const int nxt = cur ^ 1;
    if (k0 + BK < Kdim) stage(nxt, k0 + BK);   // overlap with compute below

    // Fragments from LDS (layouts per CDNA5 ISA 7.12.2):
    //   A 16x32 bf16: lane l holds row (l&15), K-chunks (l>>4)*8 and +16
    //   B 32x16 bf16: lane l holds col (l&15), K-chunks (l>>4)*8 and +16
    FragU afr[2], bfr[4];
#pragma unroll
    for (int sm = 0; sm < 2; ++sm) {
      int r = wm * 32 + sm * 16 + l16;
      int c0 = half * 8;
      afr[sm].q[0] = *(const uint4*)&As[cur][r][c0];
      afr[sm].q[1] = *(const uint4*)&As[cur][r][c0 + 16];
    }
#pragma unroll
    for (int sn = 0; sn < 4; ++sn) {
      int n = wn * 64 + sn * 16 + l16;
      int c0 = half * 8;
      bfr[sn].q[0] = *(const uint4*)&Ws[cur][n][c0];
      bfr[sn].q[1] = *(const uint4*)&Ws[cur][n][c0 + 16];
    }
#pragma unroll
    for (int sm = 0; sm < 2; ++sm)
#pragma unroll
      for (int sn = 0; sn < 4; ++sn)
        acc[sm][sn] = __builtin_amdgcn_wmma_f32_16x16x32_bf16(
            false, afr[sm].v, false, bfr[sn].v, (short)0, acc[sm][sn],
            false, false);

#if USE_ASYNC_LDS
    WAIT_ASYNC();                 // next slab landed in LDS (this wave's part)
#endif
    __syncthreads();              // everyone done reading `cur`, `nxt` visible
  }

  // Epilogue: C/D layout — VGPR r, lane l -> (M = r + 8*(l>>4), N = l&15)
#pragma unroll
  for (int sm = 0; sm < 2; ++sm)
#pragma unroll
    for (int sn = 0; sn < 4; ++sn) {
#pragma unroll
      for (int r = 0; r < 8; ++r) {
        int row = m0 + wm * 32 + sm * 16 + r + 8 * half;
        int col = n0 + wn * 64 + sn * 16 + l16;
        float v = acc[sm][sn][r];
        if (flags & GF_BIAS)   v += biasp[col];
        if (flags & GF_BIAS2)  v += bias2[col];
        if (flags & GF_ADDMAT) v += addmat[(size_t)row * ldc + col];
        if (flags & GF_RELU)   v = v > 0.0f ? v : 0.0f;
        if (outf)  outf [(size_t)row * ldc + col] = v;
        if (outbf) outbf[(size_t)row * ldc + col] = f32_to_bf16(v);
      }
    }
}

// ---------------------------------------------------------------------------
// ACT state init
// ---------------------------------------------------------------------------
__global__ void act_init_kernel(float* hp, float* rem, float* nup, float* c,
                                float* accum, unsigned short* hb, int* active,
                                unsigned int* minbits) {
  size_t i = (size_t)blockIdx.x * blockDim.x + threadIdx.x;
  if (i < (size_t)BB * DD) { c[i] = 0.0f; accum[i] = 0.0f; hb[i] = 0; }
  if (i < BB) { hp[i] = 0.0f; rem[i] = 0.0f; nup[i] = 0.0f; }
  if (i == 0) { *active = 1; *minbits = 0x7F800000u; }
}

// LSTM cell elementwise (gates already = x@Wih^T + b + h@Whh^T)
__global__ void lstm_elem_kernel(const float* __restrict__ gates,
                                 float* __restrict__ c,
                                 float* __restrict__ hnew,
                                 const int* __restrict__ active) {
  if (*active == 0) return;
  size_t idx = (size_t)blockIdx.x * blockDim.x + threadIdx.x;
  size_t b = idx / DD, d = idx % DD;
  const float* g = gates + b * FD;
  float i_ = g[d], f_ = g[DD + d], g_ = g[2 * DD + d], o_ = g[3 * DD + d];
  float cn = sigmf(f_) * c[idx] + sigmf(i_) * tanhf(g_);
  c[idx] = cn;
  hnew[idx] = sigmf(o_) * tanhf(cn);
}

// Per-token halting update: y=sig(h.halt_w+b), hp/rem/nup/acc, bf16 h for GEMM
__global__ void halt_update_kernel(const float* __restrict__ hnew,
                                   const float* __restrict__ halt_w,
                                   const float* __restrict__ halt_b,
                                   float* hp, float* rem, float* nup,
                                   float* accum, unsigned short* hb,
                                   unsigned int* minbits,
                                   const int* __restrict__ active) {
  if (*active == 0) return;
  int b = blockIdx.x;
  const float* h = hnew + (size_t)b * DD;
  __shared__ float red[256];
  float p = 0.0f;
  for (int d = threadIdx.x; d < DD; d += 256) p += h[d] * halt_w[d];
  red[threadIdx.x] = p;
  __syncthreads();
  for (int s = 128; s > 0; s >>= 1) {
    if (threadIdx.x < s) red[threadIdx.x] += red[threadIdx.x + s];
    __syncthreads();
  }
  if (threadIdx.x == 0) {
    float y   = sigmf(red[0] + halt_b[0]);
    float hpv = hp[b];
    float hpn = hpv + y * (1.0f - hpv);
    hp[b]  = hpn;
    rem[b] += 1.0f - hpn;
    nup[b] += 1.0f;
    atomicMin(minbits, __float_as_uint(hpn));   // hp >= 0 -> bit-order == fp-order
  }
  for (int d = threadIdx.x; d < DD; d += 256) {
    float v = h[d];
    accum[(size_t)b * DD + d] += v;
    hb[(size_t)b * DD + d] = f32_to_bf16(v);
  }
}

// Global early-break: active &= (min(hp) <= 0.99); reset min for next step
__global__ void step_flag_kernel(int* active, unsigned int* minbits) {
  if (*active) {
    if (__uint_as_float(*minbits) > 0.99f) *active = 0;
  }
  *minbits = 0x7F800000u;
}

// avg = acc * rem / nup  (per-token scalars broadcast over D)
__global__ void act_final_kernel(const float* __restrict__ accum,
                                 const float* __restrict__ rem,
                                 const float* __restrict__ nup,
                                 float* __restrict__ hf,
                                 unsigned short* __restrict__ hb) {
  size_t i = (size_t)blockIdx.x * blockDim.x + threadIdx.x;
  size_t b = i / DD;
  float v = accum[i] * rem[b] / nup[b];
  hf[i] = v;
  hb[i] = f32_to_bf16(v);
}

// ---------------------------------------------------------------------------
// MoE routing / grouping
// ---------------------------------------------------------------------------
__global__ void moe_init_kernel(int* counts, int* rows) {
  int i = blockIdx.x * blockDim.x + threadIdx.x;
  if (i < NE) counts[i] = 0;
  if (i < CAP) rows[i] = -1;
}

__global__ void route_kernel(const float* __restrict__ hf,
                             const float* __restrict__ gw,
                             const float* __restrict__ gb,
                             float* __restrict__ wt, int* __restrict__ eidx,
                             int* __restrict__ counts) {
  int b = blockIdx.x;
  const float* x = hf + (size_t)b * DD;
  float p[NE];
#pragma unroll
  for (int e = 0; e < NE; ++e) p[e] = 0.0f;
  for (int d = threadIdx.x; d < DD; d += 256) {
    float xv = x[d];
#pragma unroll
    for (int e = 0; e < NE; ++e) p[e] += xv * gw[(size_t)e * DD + d];
  }
  __shared__ float red[256];
  __shared__ float logits[NE];
  for (int e = 0; e < NE; ++e) {
    red[threadIdx.x] = p[e];
    __syncthreads();
    for (int s = 128; s > 0; s >>= 1) {
      if (threadIdx.x < s) red[threadIdx.x] += red[threadIdx.x + s];
      __syncthreads();
    }
    if (threadIdx.x == 0) logits[e] = red[0] + gb[e];
    __syncthreads();
  }
  if (threadIdx.x == 0) {
    int i0 = 0; float m0 = logits[0];
    for (int e = 1; e < NE; ++e) if (logits[e] > m0) { m0 = logits[e]; i0 = e; }
    int i1 = -1; float m1 = -3.4e38f;
    for (int e = 0; e < NE; ++e)
      if (e != i0 && logits[e] > m1) { m1 = logits[e]; i1 = e; }
    float e1 = __expf(m1 - m0);
    float s = 1.0f + e1;
    wt[b * 2 + 0] = 1.0f / s;
    wt[b * 2 + 1] = e1 / s;
    eidx[b * 2 + 0] = i0;
    eidx[b * 2 + 1] = i1;
    atomicAdd(&counts[i0], 1);
    atomicAdd(&counts[i1], 1);
  }
}

__global__ void offsets_kernel(const int* counts, int* offs, int* cursors) {
  int o = 0;
  for (int e = 0; e < NE; ++e) {
    offs[e] = o;
    cursors[e] = o;
    o += (counts[e] + (BM - 1)) & ~(BM - 1);   // 128-align so tiles never straddle
  }
  offs[NE] = o;
}

__global__ void scatter_kernel(const int* __restrict__ eidx, int* cursors,
                               int* __restrict__ rows, int* __restrict__ posmap) {
  int t = blockIdx.x * blockDim.x + threadIdx.x;
  if (t >= BB * KTOP) return;
  int e = eidx[t];
  int r = atomicAdd(&cursors[e], 1);
  rows[r] = t >> 1;            // token id
  posmap[t] = r;
}

__global__ void gather_kernel(const int* __restrict__ rows,
                              const unsigned short* __restrict__ hb,
                              unsigned short* __restrict__ Ag) {
  int r = blockIdx.x;
  int src = rows[r];
  for (int d = threadIdx.x; d < DD; d += 256)
    Ag[(size_t)r * DD + d] = (src >= 0) ? hb[(size_t)src * DD + d]
                                        : (unsigned short)0;
}

__global__ void combine_kernel(const float* __restrict__ wt,
                               const int* __restrict__ posmap,
                               const float* __restrict__ Og,
                               float* __restrict__ hf,
                               unsigned short* __restrict__ hb) {
  int b = blockIdx.x;
  float w0 = wt[b * 2 + 0], w1 = wt[b * 2 + 1];
  size_t r0 = (size_t)posmap[b * 2 + 0] * DD;
  size_t r1 = (size_t)posmap[b * 2 + 1] * DD;
  for (int d = threadIdx.x; d < DD; d += 256) {
    float v = w0 * Og[r0 + d] + w1 * Og[r1 + d];
    hf[(size_t)b * DD + d] = v;
    hb[(size_t)b * DD + d] = f32_to_bf16(v);
  }
}

// ---------------------------------------------------------------------------
// Final LayerNorm -> d_out (fp32)
// ---------------------------------------------------------------------------
__global__ void layernorm_kernel(const float* __restrict__ hf,
                                 const float* __restrict__ g,
                                 const float* __restrict__ bta,
                                 float* __restrict__ out) {
  int b = blockIdx.x;
  const float* x = hf + (size_t)b * DD;
  __shared__ float rs[256], rs2[256];
  float s = 0.0f, s2 = 0.0f;
  for (int d = threadIdx.x; d < DD; d += 256) {
    float v = x[d];
    s += v; s2 += v * v;
  }
  rs[threadIdx.x] = s; rs2[threadIdx.x] = s2;
  __syncthreads();
  for (int st = 128; st > 0; st >>= 1) {
    if (threadIdx.x < st) {
      rs[threadIdx.x]  += rs[threadIdx.x + st];
      rs2[threadIdx.x] += rs2[threadIdx.x + st];
    }
    __syncthreads();
  }
  float mean = rs[0] / DD;
  float var  = rs2[0] / DD - mean * mean;
  float rstd = rsqrtf(var + 1e-5f);
  for (int d = threadIdx.x; d < DD; d += 256)
    out[(size_t)b * DD + d] = (x[d] - mean) * rstd * g[d] + bta[d];
}

// ---------------------------------------------------------------------------
extern "C" void kernel_launch(void* const* d_in, const int* in_sizes, int n_in,
                              void* d_out, int out_size, void* d_ws,
                              size_t ws_size, hipStream_t stream) {
  const float* x      = (const float*)d_in[0];
  const float* w_ih   = (const float*)d_in[1];
  const float* w_hh   = (const float*)d_in[2];
  const float* b_ih   = (const float*)d_in[3];
  const float* b_hh   = (const float*)d_in[4];
  const float* halt_w = (const float*)d_in[5];
  const float* halt_b = (const float*)d_in[6];
  const float* gate_w = (const float*)d_in[7];
  const float* gate_b = (const float*)d_in[8];
  const float* w1     = (const float*)d_in[9];
  const float* b1     = (const float*)d_in[10];
  const float* w2     = (const float*)d_in[11];
  const float* b2     = (const float*)d_in[12];
  const float* ln_g   = (const float*)d_in[13];
  const float* ln_b   = (const float*)d_in[14];

  // ---- workspace carve (~1 GB) ----
  char* base = (char*)d_ws;
  size_t off = 0;
  auto carve = [&](size_t bytes) -> void* {
    void* p = base + off;
    off = (off + bytes + 255) & ~(size_t)255;
    return p;
  };
  unsigned short* wihb = (unsigned short*)carve((size_t)FD * DD * 2);
  unsigned short* whhb = (unsigned short*)carve((size_t)FD * DD * 2);
  unsigned short* w1b  = (unsigned short*)carve((size_t)NL * NE * DD * DD * 2);
  unsigned short* w2b  = (unsigned short*)carve((size_t)NL * NE * DD * DD * 2);
  unsigned short* xb   = (unsigned short*)carve((size_t)BB * DD * 2);
  unsigned short* hb   = (unsigned short*)carve((size_t)BB * DD * 2);
  unsigned short* Ag   = (unsigned short*)carve((size_t)CAP * DD * 2);
  unsigned short* Hg   = (unsigned short*)carve((size_t)CAP * DD * 2);
  float* G0    = (float*)carve((size_t)BB * FD * 4);
  float* gates = (float*)carve((size_t)BB * FD * 4);
  float* cst   = (float*)carve((size_t)BB * DD * 4);
  float* hnew  = (float*)carve((size_t)BB * DD * 4);
  float* accum = (float*)carve((size_t)BB * DD * 4);
  float* hf    = (float*)carve((size_t)BB * DD * 4);
  float* Og    = (float*)carve((size_t)CAP * DD * 4);
  float* hp    = (float*)carve((size_t)BB * 4);
  float* rem   = (float*)carve((size_t)BB * 4);
  float* nup   = (float*)carve((size_t)BB * 4);
  float* wt    = (float*)carve((size_t)BB * KTOP * 4);
  int* eidx    = (int*)carve((size_t)BB * KTOP * 4);
  int* counts  = (int*)carve(NE * 4);
  int* cursors = (int*)carve(NE * 4);
  int* offs    = (int*)carve((NE + 1) * 4);
  int* rows    = (int*)carve(CAP * 4);
  int* posmap  = (int*)carve((size_t)BB * KTOP * 4);
  int* active  = (int*)carve(4);
  unsigned int* minbits = (unsigned int*)carve(4);
  (void)ws_size; (void)n_in; (void)in_sizes; (void)out_size;

  // ---- fp32 -> bf16 weight/activation conversion ----
  cvt_bf16_kernel<<<4096, 256, 0, stream>>>(w_ih, wihb, (size_t)FD * DD);
  cvt_bf16_kernel<<<4096, 256, 0, stream>>>(w_hh, whhb, (size_t)FD * DD);
  cvt_bf16_kernel<<<8192, 256, 0, stream>>>(w1, w1b, (size_t)NL * NE * DD * DD);
  cvt_bf16_kernel<<<8192, 256, 0, stream>>>(w2, w2b, (size_t)NL * NE * DD * DD);
  cvt_bf16_kernel<<<2048, 256, 0, stream>>>(x, xb, (size_t)BB * DD);

  act_init_kernel<<<(BB * DD) / 256, 256, 0, stream>>>(hp, rem, nup, cst, accum,
                                                       hb, active, minbits);

  // ---- G0 = x @ w_ih^T + (b_ih + b_hh) : loop-invariant part of the gates ----
  dim3 gLstm(BB / BM, FD / BN);
  gemm_bf16_kernel<<<gLstm, 256, 0, stream>>>(
      xb, DD, wihb, G0, nullptr, FD, b_ih, b_hh, nullptr, DD,
      GF_BIAS | GF_BIAS2, nullptr, nullptr, 0, 0);

  // ---- ACT loop (globally gated by `active`, exact early-break semantics) ----
  for (int t = 0; t < NSTEP; ++t) {
    gemm_bf16_kernel<<<gLstm, 256, 0, stream>>>(
        hb, DD, whhb, gates, nullptr, FD, nullptr, nullptr, G0, DD,
        GF_ADDMAT, active, nullptr, 0, 0);
    lstm_elem_kernel<<<(BB * DD) / 256, 256, 0, stream>>>(gates, cst, hnew,
                                                          active);
    halt_update_kernel<<<BB, 256, 0, stream>>>(hnew, halt_w, halt_b, hp, rem,
                                               nup, accum, hb, minbits, active);
    step_flag_kernel<<<1, 1, 0, stream>>>(active, minbits);
  }
  act_final_kernel<<<(BB * DD) / 256, 256, 0, stream>>>(accum, rem, nup, hf, hb);

  // ---- MoE levels: route -> group -> expert GEMMs (top-2 only) -> combine ----
  for (int lvl = 0; lvl < NL; ++lvl) {
    moe_init_kernel<<<(CAP + 255) / 256, 256, 0, stream>>>(counts, rows);
    route_kernel<<<BB, 256, 0, stream>>>(hf, gate_w + (size_t)lvl * NE * DD,
                                         gate_b + (size_t)lvl * NE, wt, eidx,
                                         counts);
    offsets_kernel<<<1, 1, 0, stream>>>(counts, offs, cursors);
    scatter_kernel<<<(BB * KTOP) / 256, 256, 0, stream>>>(eidx, cursors, rows,
                                                          posmap);
    gather_kernel<<<CAP, 256, 0, stream>>>(rows, hb, Ag);

    dim3 gMoe(CAP / BM, DD / BN);
    // Hg = relu(Ag @ w1[e]^T + b1[e])  (bf16 out only)
    gemm_bf16_kernel<<<gMoe, 256, 0, stream>>>(
        Ag, DD, w1b + (size_t)lvl * NE * DD * DD, nullptr, Hg, DD,
        b1 + (size_t)lvl * NE * DD, nullptr, nullptr, DD, GF_BIAS | GF_RELU,
        nullptr, offs, DD * DD, DD);
    // Og = Hg @ w2[e]^T + b2[e]  (f32 out)
    gemm_bf16_kernel<<<gMoe, 256, 0, stream>>>(
        Hg, DD, w2b + (size_t)lvl * NE * DD * DD, Og, nullptr, DD,
        b2 + (size_t)lvl * NE * DD, nullptr, nullptr, DD, GF_BIAS,
        nullptr, offs, DD * DD, DD);
    combine_kernel<<<BB, 256, 0, stream>>>(wt, posmap, Og, hf, hb);
  }

  layernorm_kernel<<<BB, 256, 0, stream>>>(hf, ln_g, ln_b, (float*)d_out);
}
